// PointPatchEmbed_31636729102601
// MI455X (gfx1250) — matrix operand order repaired and checked by reading, not compiled, not run on hardware
//
#include <hip/hip_runtime.h>
#include <hip/hip_bf16.h>

// ---------------- problem constants ----------------
#define BB 16
#define NN 8192
#define GG 512
#define KK 32
#define DD 384
#define C1 128
#define C2 256
#define MROWS (BB*GG*KK)      // 262144 rows through the MLP
#define ROWS 128              // rows per block in GEMM kernels (4 groups)
#define NBLK (MROWS/ROWS)     // 2048

typedef __attribute__((ext_vector_type(16))) _Float16 v16h;
typedef __attribute__((ext_vector_type(8)))  _Float16 v8h;
typedef __attribute__((ext_vector_type(8)))  float    v8f;

__device__ __forceinline__ v8f wmma_f16(v16h a, v16h b, v8f c) {
  return __builtin_amdgcn_wmma_f32_16x16x32_f16(false, a, false, b, (short)0, c, false, false);
}

// A-tile (16x32 f16) from row-major LDS [row][ldk]:
// lanes 0-15: rows 0-15, K 0-7 / 16-23 ; lanes 16-31: same rows, K 8-15 / 24-31
__device__ __forceinline__ v16h load_a(const _Float16* a, int ldk, int mtile, int ks) {
  int lane = threadIdx.x & 31;
  int half = lane >> 4;
  int row  = (mtile << 4) + (lane & 15);
  const _Float16* base = a + row * ldk + ks * 32 + half * 8;
  union { v16h v; v8h h[2]; } u;
  u.h[0] = *(const v8h*)(base);
  u.h[1] = *(const v8h*)(base + 16);
  return u.v;
}

// B-tile (32x16 f16) from row-major weights W[d][k]:
// lane holds output channel d = ntile*16 + lane%16, K = (lane/16)*8 .. +7 and 16+(lane/16)*8 .. +7 within kstep
__device__ __forceinline__ v16h load_b(const _Float16* W, int ldk, int ntile, int ks) {
  int lane = threadIdx.x & 31;
  int half = lane >> 4;
  int dch  = (ntile << 4) + (lane & 15);
  const _Float16* base = W + dch * ldk + ks * 32 + half * 8;
  union { v16h v; v8h h[2]; } u;
  u.h[0] = *(const v8h*)(base);
  u.h[1] = *(const v8h*)(base + 16);
  return u.v;
}

// ---------------- FPS: one block per batch ----------------
__global__ void fps_kernel(const float* __restrict__ pc, float* __restrict__ centers_ws,
                           float* __restrict__ out_centers) {
  extern __shared__ float sm[];
  float* px = sm;            // N
  float* py = sm + NN;       // N
  float* pz = sm + 2 * NN;   // N
  float* dist = sm + 3 * NN; // N
  float* rv = sm + 4 * NN;   // 1024
  int*   ri = (int*)(rv + 1024);
  const int b = blockIdx.x;
  const float* p = pc + (size_t)b * NN * 3;
  for (int i = threadIdx.x; i < NN; i += 1024) {
    px[i] = p[i * 3 + 0]; py[i] = p[i * 3 + 1]; pz[i] = p[i * 3 + 2];
    dist[i] = 3.4e38f;
  }
  __syncthreads();
  int last = 0;
  for (int it = 0; it < GG; ++it) {
    float cx = px[last], cy = py[last], cz = pz[last];
    if (threadIdx.x == 0) {
      size_t o = ((size_t)b * GG + it) * 3;
      centers_ws[o] = cx; centers_ws[o + 1] = cy; centers_ws[o + 2] = cz;
      out_centers[o] = cx; out_centers[o + 1] = cy; out_centers[o + 2] = cz;
    }
    float bestv = -1.f; int besti = 0;
    for (int i = threadIdx.x; i < NN; i += 1024) {
      float dx = px[i] - cx, dy = py[i] - cy, dz = pz[i] - cz;
      float d = dx * dx + dy * dy + dz * dz;
      float dd = dist[i];
      dd = d < dd ? d : dd;
      dist[i] = dd;
      if (dd > bestv) { bestv = dd; besti = i; }
    }
    rv[threadIdx.x] = bestv; ri[threadIdx.x] = besti;
    __syncthreads();
    for (int s = 512; s > 0; s >>= 1) {
      if (threadIdx.x < s) {
        float ov = rv[threadIdx.x + s]; int oi = ri[threadIdx.x + s];
        float mv = rv[threadIdx.x];     int mi = ri[threadIdx.x];
        if (ov > mv || (ov == mv && oi < mi)) { rv[threadIdx.x] = ov; ri[threadIdx.x] = oi; }
      }
      __syncthreads();
    }
    last = ri[0];
    __syncthreads();
  }
}

// ---------------- KNN + group + center-subtract: one block per (b,g) ----------------
__global__ void knn_kernel(const float* __restrict__ pc, const float* __restrict__ centers,
                           float* __restrict__ grouped) {
  extern __shared__ float sm[];
  float* d2 = sm;           // N
  float* rv = sm + NN;      // 256
  int*   ri = (int*)(rv + 256);
  int*   sel = ri + 256;    // K
  const int bg = blockIdx.x;
  const int b = bg / GG;
  const float* p = pc + (size_t)b * NN * 3;
  const float cx = centers[bg * 3], cy = centers[bg * 3 + 1], cz = centers[bg * 3 + 2];
  for (int i = threadIdx.x; i < NN; i += 256) {
    float dx = p[i * 3] - cx, dy = p[i * 3 + 1] - cy, dz = p[i * 3 + 2] - cz;
    d2[i] = dx * dx + dy * dy + dz * dz;
  }
  __syncthreads();
  for (int k = 0; k < KK; ++k) {
    float bv = 3.4e38f; int bi = NN;
    for (int i = threadIdx.x; i < NN; i += 256) {
      float v = d2[i];
      if (v < bv || (v == bv && i < bi)) { bv = v; bi = i; }
    }
    rv[threadIdx.x] = bv; ri[threadIdx.x] = bi;
    __syncthreads();
    for (int s = 128; s > 0; s >>= 1) {
      if (threadIdx.x < s) {
        float ov = rv[threadIdx.x + s]; int oi = ri[threadIdx.x + s];
        float mv = rv[threadIdx.x];     int mi = ri[threadIdx.x];
        if (ov < mv || (ov == mv && oi < mi)) { rv[threadIdx.x] = ov; ri[threadIdx.x] = oi; }
      }
      __syncthreads();
    }
    if (threadIdx.x == 0) { sel[k] = ri[0]; d2[ri[0]] = 3.4e38f; }
    __syncthreads();
  }
  if (threadIdx.x < KK * 3) {
    int k = threadIdx.x / 3, c = threadIdx.x % 3;
    int pi = sel[k];
    float cv = (c == 0 ? cx : (c == 1 ? cy : cz));
    grouped[((size_t)bg * KK + k) * 3 + c] = p[pi * 3 + c] - cv;
  }
}

// ---------------- moments of grouped for analytic layer-1 BN stats ----------------
__global__ void s1_partial(const float* __restrict__ grouped, float* __restrict__ part) {
  float a[9] = {0, 0, 0, 0, 0, 0, 0, 0, 0};
  for (size_t r = (size_t)blockIdx.x * 256 + threadIdx.x; r < (size_t)MROWS; r += 256 * 256) {
    float x = grouped[r * 3], y = grouped[r * 3 + 1], z = grouped[r * 3 + 2];
    a[0] += x; a[1] += y; a[2] += z;
    a[3] += x * x; a[4] += x * y; a[5] += x * z;
    a[6] += y * y; a[7] += y * z; a[8] += z * z;
  }
  __shared__ float red[256];
  for (int c = 0; c < 9; ++c) {
    red[threadIdx.x] = a[c];
    __syncthreads();
    for (int s = 128; s > 0; s >>= 1) {
      if (threadIdx.x < s) red[threadIdx.x] += red[threadIdx.x + s];
      __syncthreads();
    }
    if (threadIdx.x == 0) part[blockIdx.x * 12 + c] = red[0];
    __syncthreads();
  }
}

__global__ void s1_final(const float* __restrict__ part, const float* __restrict__ W1,
                         const float* __restrict__ b1, const float* __restrict__ g1,
                         const float* __restrict__ be1, float* __restrict__ sc1) {
  __shared__ float mom[9];
  if (threadIdx.x < 9) {
    float s = 0.f;
    for (int p = 0; p < 256; ++p) s += part[p * 12 + threadIdx.x];
    mom[threadIdx.x] = s * (1.0f / (float)MROWS);
  }
  __syncthreads();
  int d = threadIdx.x;
  if (d < C1) {
    float w0 = W1[d * 3], w1 = W1[d * 3 + 1], w2 = W1[d * 3 + 2];
    float dotE = w0 * mom[0] + w1 * mom[1] + w2 * mom[2];
    float mu = dotE + b1[d];
    float q = w0 * w0 * mom[3] + w1 * w1 * mom[6] + w2 * w2 * mom[8]
            + 2.f * (w0 * w1 * mom[4] + w0 * w2 * mom[5] + w1 * w2 * mom[7]);
    float m2 = q + 2.f * b1[d] * dotE + b1[d] * b1[d];
    float var = m2 - mu * mu;
    float rstd = rsqrtf(var + 1e-5f);
    float sc = g1[d] * rstd;
    sc1[d] = sc;                              // scale (applied to raw dot)
    sc1[C1 + d] = (b1[d] - mu) * sc + be1[d]; // fused bias
  }
}

// ---------------- weights f32 -> f16 ----------------
__global__ void wconv(const float* __restrict__ W2, const float* __restrict__ W3,
                      _Float16* __restrict__ W2h, _Float16* __restrict__ W3h) {
  int i = blockIdx.x * 256 + threadIdx.x;
  if (i < C2 * C1) W2h[i] = (_Float16)W2[i];
  if (i < DD * C2) W3h[i] = (_Float16)W3[i];
}

// build BN1+ReLU'ed 128-wide activations for ROWS rows into LDS (f16, row-major)
__device__ __forceinline__ void build_a1(const float* __restrict__ grouped, size_t row0,
                                         const float* __restrict__ W1, const float* __restrict__ sc1,
                                         _Float16* a1, float* gstage, float* w1s, float* s1s) {
  int tid = threadIdx.x;
  for (int i = tid; i < ROWS * 3; i += 256) gstage[i] = grouped[row0 * 3 + i];
  for (int i = tid; i < C1 * 3; i += 256) w1s[i] = W1[i];
  for (int i = tid; i < 2 * C1; i += 256) s1s[i] = sc1[i];
  __syncthreads();
  for (int i = tid; i < ROWS * C1; i += 256) {
    int r = i >> 7, d = i & 127;
    float dot = gstage[r * 3] * w1s[d * 3] + gstage[r * 3 + 1] * w1s[d * 3 + 1]
              + gstage[r * 3 + 2] * w1s[d * 3 + 2];
    float v = dot * s1s[d] + s1s[C1 + d];
    v = v > 0.f ? v : 0.f;
    a1[r * C1 + d] = (_Float16)v;
  }
  __syncthreads();
}

// ---------------- layer-2 stats pass (WMMA), deterministic partials ----------------
__global__ void s2_partial(const float* __restrict__ grouped, const float* __restrict__ W1,
                           const float* __restrict__ sc1, const _Float16* __restrict__ W2h,
                           float* __restrict__ part) {
  extern __shared__ char smraw[];
  _Float16* a1 = (_Float16*)smraw;                       // ROWS*C1 f16
  float* gstage = (float*)(smraw + ROWS * C1 * 2);       // ROWS*3
  float* w1s = gstage + ROWS * 3;                        // 384
  float* s1s = w1s + C1 * 3;                             // 256
  float* psum = s1s + 2 * C1;                            // 2*256
  float* pq = psum + 2 * C2;                             // 2*256
  size_t row0 = (size_t)blockIdx.x * ROWS;
  build_a1(grouped, row0, W1, sc1, a1, gstage, w1s, s1s);
  int wave = threadIdx.x >> 5, lane = threadIdx.x & 31, half = lane >> 4, ln = lane & 15;
  for (int t = 0; t < 2; ++t) {
    int ntile = wave + t * 8;
    v16h bmat[4];
    for (int ks = 0; ks < 4; ++ks) bmat[ks] = load_b(W2h, C1, ntile, ks);
    float s = 0.f, q = 0.f;
    for (int mt = 0; mt < 8; ++mt) {
      v8f acc = {};
      for (int ks = 0; ks < 4; ++ks) acc = wmma_f16(load_a(a1, C1, mt, ks), bmat[ks], acc);
      for (int r = 0; r < 8; ++r) { float v = acc[r]; s += v; q += v * v; }
    }
    psum[half * C2 + ntile * 16 + ln] = s;
    pq[half * C2 + ntile * 16 + ln] = q;
  }
  __syncthreads();
  for (int d = threadIdx.x; d < C2; d += 256) {
    part[(size_t)blockIdx.x * 512 + d] = psum[d] + psum[C2 + d];
    part[(size_t)blockIdx.x * 512 + C2 + d] = pq[d] + pq[C2 + d];
  }
}

__global__ void s2_final(const float* __restrict__ part, const float* __restrict__ b2,
                         const float* __restrict__ g2, const float* __restrict__ be2,
                         float* __restrict__ sc2) {
  int d = threadIdx.x; // 256 threads
  float s = 0.f, q = 0.f;
  for (int p = 0; p < NBLK; ++p) {
    s += part[(size_t)p * 512 + d];
    q += part[(size_t)p * 512 + C2 + d];
  }
  float inv = 1.0f / (float)MROWS;
  float ea = s * inv, ea2 = q * inv;
  float mu = ea + b2[d];
  float m2 = ea2 + 2.f * b2[d] * ea + b2[d] * b2[d];
  float var = m2 - mu * mu;
  float rstd = rsqrtf(var + 1e-5f);
  float sc = g2[d] * rstd;
  sc2[d] = sc;                              // scale on raw accumulator
  sc2[C2 + d] = (b2[d] - mu) * sc + be2[d]; // fused bias
}

// ---------------- fused GEMM2 -> BN2+ReLU -> GEMM3 -> +b3 -> max over K ----------------
__global__ void final_fused(const float* __restrict__ grouped, const float* __restrict__ W1,
                            const float* __restrict__ sc1, const _Float16* __restrict__ W2h,
                            const _Float16* __restrict__ W3h, const float* __restrict__ sc2,
                            const float* __restrict__ b3, float* __restrict__ out) {
  extern __shared__ char smraw[];
  _Float16* a1 = (_Float16*)smraw;                            // ROWS*128 f16 (32KB)
  _Float16* a2 = (_Float16*)(smraw + ROWS * C1 * 2);          // ROWS*256 f16 (64KB)
  char* pp = smraw + ROWS * C1 * 2 + ROWS * C2 * 2;
  float* gstage = (float*)pp;      pp += ROWS * 3 * 4;
  float* w1s = (float*)pp;         pp += C1 * 3 * 4;
  float* s1s = (float*)pp;         pp += 2 * C1 * 4;
  float* s2s = (float*)pp;         pp += 2 * C2 * 4;
  float* b3s = (float*)pp;         pp += DD * 4;
  int tid = threadIdx.x;
  for (int i = tid; i < 2 * C2; i += 256) s2s[i] = sc2[i];
  for (int i = tid; i < DD; i += 256) b3s[i] = b3[i];
  size_t row0 = (size_t)blockIdx.x * ROWS;
  build_a1(grouped, row0, W1, sc1, a1, gstage, w1s, s1s);
  int wave = tid >> 5, lane = tid & 31, half = lane >> 4, ln = lane & 15;
  // ---- layer 2 into a2 (BN2 + ReLU, f16) ----
  for (int t = 0; t < 2; ++t) {
    int ntile = wave + t * 8;
    int d = ntile * 16 + ln;
    float sc = s2s[d], bi = s2s[C2 + d];
    v16h bmat[4];
    for (int ks = 0; ks < 4; ++ks) bmat[ks] = load_b(W2h, C1, ntile, ks);
    for (int mt = 0; mt < 8; ++mt) {
      v8f acc = {};
      for (int ks = 0; ks < 4; ++ks) acc = wmma_f16(load_a(a1, C1, mt, ks), bmat[ks], acc);
      for (int r = 0; r < 8; ++r) {
        float v = acc[r] * sc + bi;
        v = v > 0.f ? v : 0.f;
        int m = mt * 16 + half * 8 + r;
        a2[m * C2 + d] = (_Float16)v;
      }
    }
  }
  __syncthreads();
  // ---- layer 3 + max over each group of 32 rows ----
  for (int t = 0; t < 3; ++t) {
    int ntile = wave + t * 8;
    int d = ntile * 16 + ln;
    float bb = b3s[d];
    v16h bmat[8];
    for (int ks = 0; ks < 8; ++ks) bmat[ks] = load_b(W3h, C2, ntile, ks);
    float gmax = -3.4e38f;
    for (int mt = 0; mt < 8; ++mt) {
      v8f acc = {};
      for (int ks = 0; ks < 8; ++ks) acc = wmma_f16(load_a(a2, C2, mt, ks), bmat[ks], acc);
      float lm = -3.4e38f;
      for (int r = 0; r < 8; ++r) { float v = acc[r] + bb; lm = v > lm ? v : lm; }
      if ((mt & 1) == 0) {
        gmax = lm;
      } else {
        gmax = gmax > lm ? gmax : lm;
        float o = __shfl_xor(gmax, 16, 32);      // combine halves (wave32)
        gmax = gmax > o ? gmax : o;
        if (half == 0) {
          int g = mt >> 1;
          out[((size_t)blockIdx.x * 4 + g) * DD + d] = gmax;
        }
      }
    }
  }
}

extern "C" void kernel_launch(void* const* d_in, const int* in_sizes, int n_in,
                              void* d_out, int out_size, void* d_ws, size_t ws_size,
                              hipStream_t stream) {
  (void)in_sizes; (void)n_in; (void)out_size; (void)ws_size;
  const float* pc  = (const float*)d_in[0];
  const float* W1  = (const float*)d_in[1];
  const float* b1  = (const float*)d_in[2];
  const float* g1  = (const float*)d_in[3];
  const float* be1 = (const float*)d_in[4];
  const float* W2  = (const float*)d_in[5];
  const float* b2  = (const float*)d_in[6];
  const float* g2  = (const float*)d_in[7];
  const float* be2 = (const float*)d_in[8];
  const float* W3  = (const float*)d_in[9];
  const float* b3  = (const float*)d_in[10];
  float* out = (float*)d_out;
  float* out_centers = out + (size_t)BB * GG * DD;

  char* ws = (char*)d_ws;
  auto alloc = [&](size_t bytes) -> char* {
    char* r = ws;
    ws += (bytes + 255) & ~(size_t)255;
    return r;
  };
  float*    centers = (float*)alloc((size_t)BB * GG * 3 * 4);
  float*    grouped = (float*)alloc((size_t)MROWS * 3 * 4);
  _Float16* W2h     = (_Float16*)alloc((size_t)C2 * C1 * 2);
  _Float16* W3h     = (_Float16*)alloc((size_t)DD * C2 * 2);
  float*    s1p     = (float*)alloc((size_t)256 * 12 * 4);
  float*    sc1     = (float*)alloc((size_t)2 * C1 * 4);
  float*    s2p     = (float*)alloc((size_t)NBLK * 512 * 4);
  float*    sc2     = (float*)alloc((size_t)2 * C2 * 4);

  size_t fps_lds = ((size_t)4 * NN + 2 * 1024) * 4;
  fps_kernel<<<BB, 1024, fps_lds, stream>>>(pc, centers, out_centers);

  size_t knn_lds = ((size_t)NN + 256 + 256 + KK) * 4;
  knn_kernel<<<BB * GG, 256, knn_lds, stream>>>(pc, centers, grouped);

  wconv<<<(DD * C2 + 255) / 256, 256, 0, stream>>>(W2, W3, W2h, W3h);

  s1_partial<<<256, 256, 0, stream>>>(grouped, s1p);
  s1_final<<<1, 128, 0, stream>>>(s1p, W1, b1, g1, be1, sc1);

  size_t s2_lds = (size_t)ROWS * C1 * 2 + ((size_t)ROWS * 3 + C1 * 3 + 2 * C1 + 4 * C2) * 4;
  s2_partial<<<NBLK, 256, s2_lds, stream>>>(grouped, W1, sc1, W2h, s2p);
  s2_final<<<1, 256, 0, stream>>>(s2p, b2, g2, be2, sc2);

  size_t f_lds = (size_t)ROWS * C1 * 2 + (size_t)ROWS * C2 * 2
               + ((size_t)ROWS * 3 + C1 * 3 + 2 * C1 + 2 * C2 + DD) * 4;
  final_fused<<<NBLK, 256, f_lds, stream>>>(grouped, W1, sc1, W2h, W3h, sc2, b3, out);
}
